// MultiHeadAttentionLayer_53034256171293
// MI455X (gfx1250) — matrix operand order repaired and checked by reading, not compiled, not run on hardware
//
#include <hip/hip_runtime.h>
#include <math.h>

// ---------------------------------------------------------------------------
// MultiHeadAttentionLayer for MI455X (gfx1250, wave32, WMMA).
//
// M = bs*N = 16384 nodes, C = 256, H = 8, D = 64, H*D = 512, E = 524288.
//
// The torch-faithful double reshape cancels in memory:
//   Qn[i,h,d] == RoPE( (X @ Wq^T)[i, h*64+d] ),  RoPE position index
//   n2 = (i % 512)*8 + h  in batch b = i/4096, axis = (d<32 ? x : y),
//   freq j = d % 16.
//
// GEMM strategy: bf16x3 split (x = hi + lo; x*w ~= hi*wh + hi*wl + lo*wh)
// gives ~fp32 accuracy on the 16x16x32 bf16 WMMA pipe (8x MACs/instr vs the
// f32 16x16x4 pipe).  The fp32->bf16 split is done ONCE by pre-pack kernels
// that emit operands in exact WMMA fragment order (contiguous 32B per lane),
// so the GEMM hot loop is pure {global_load_b128 + v_wmma}.
// ---------------------------------------------------------------------------

typedef __attribute__((ext_vector_type(16))) __bf16 v16bf;
typedef __attribute__((ext_vector_type(8)))  float  v8f;

#define CIN   256
#define HEADS 8
#define DHEAD 64
#define COUT  512   // HEADS*DHEAD
#define KT    8     // CIN/32 k-tiles
#define NT    32    // COUT/16 n-tiles

// ---------------------------------------------------------------------------
// Pack X (M,256) f32 -> hi/lo bf16 in A-fragment order.
// A frag (16x32 bf16): lane = 16*hi16half K-grouping; element e:
//   e in [0,8)  -> K = kt*32 + hi16*8 + e
//   e in [8,16) -> K = kt*32 + hi16*8 + 8 + e   (i.e. offsets 16..23)
// Linear index: ((rt*KT + kt)*32 + lane) * 16 elements.
// ---------------------------------------------------------------------------
__global__ __launch_bounds__(256)
void pack_x_kernel(const float* __restrict__ X,
                   __bf16* __restrict__ Xh, __bf16* __restrict__ Xl, int M)
{
    const int idx = blockIdx.x * 256 + threadIdx.x;
    if (idx >= (M / 16) * KT * 32) return;
    const int lane = idx & 31;
    const int kt   = (idx >> 5) & (KT - 1);
    const int rt   = idx >> 8;
    const int r    = lane & 15;
    const int hi16 = lane >> 4;

    const float* __restrict__ Xrow =
        X + (size_t)(rt * 16 + r) * CIN + kt * 32 + hi16 * 8;
    float4 a0 = *(const float4*)(Xrow + 0);
    float4 a1 = *(const float4*)(Xrow + 4);
    float4 a2 = *(const float4*)(Xrow + 16);
    float4 a3 = *(const float4*)(Xrow + 20);
    float xs[16] = {a0.x, a0.y, a0.z, a0.w, a1.x, a1.y, a1.z, a1.w,
                    a2.x, a2.y, a2.z, a2.w, a3.x, a3.y, a3.z, a3.w};
    v16bf vh, vl;
    #pragma unroll
    for (int e = 0; e < 16; ++e) {
        __bf16 hi = (__bf16)xs[e];
        vh[e] = hi;
        vl[e] = (__bf16)(xs[e] - (float)hi);
    }
    ((v16bf*)Xh)[idx] = vh;
    ((v16bf*)Xl)[idx] = vl;
}

// ---------------------------------------------------------------------------
// Pack Wq/Wk/Wv (512,256) f32 -> hi/lo bf16 in B-fragment order.
// B frag (32x16 bf16): lane = column (out-feature); element e ->
// K = kt*32 + hi16*8 + e (e<8), kt*32 + 16 + hi16*8 + (e-8) (e>=8); B[k][n] = W[n][k].
// Linear index: (((m*NT + nt)*KT + kt)*32 + lane) * 16 elements.
// ---------------------------------------------------------------------------
__global__ __launch_bounds__(256)
void pack_w_kernel(const float* __restrict__ Wq, const float* __restrict__ Wk,
                   const float* __restrict__ Wv,
                   __bf16* __restrict__ Wh, __bf16* __restrict__ Wl)
{
    const int idx = blockIdx.x * 256 + threadIdx.x;
    if (idx >= 3 * NT * KT * 32) return;
    const int lane = idx & 31;
    const int kt   = (idx >> 5) & (KT - 1);
    const int nt   = (idx >> 8) & (NT - 1);
    const int m    = idx >> 13;
    const float* __restrict__ W = (m == 0) ? Wq : (m == 1) ? Wk : Wv;
    const int r    = lane & 15;
    const int hi16 = lane >> 4;

    const float* __restrict__ Wrow =
        W + (size_t)(nt * 16 + r) * CIN + kt * 32 + hi16 * 8;
    float4 b0 = *(const float4*)(Wrow + 0);
    float4 b1 = *(const float4*)(Wrow + 4);
    float4 b2 = *(const float4*)(Wrow + 16);
    float4 b3 = *(const float4*)(Wrow + 20);
    float xs[16] = {b0.x, b0.y, b0.z, b0.w, b1.x, b1.y, b1.z, b1.w,
                    b2.x, b2.y, b2.z, b2.w, b3.x, b3.y, b3.z, b3.w};
    v16bf vh, vl;
    #pragma unroll
    for (int e = 0; e < 16; ++e) {
        __bf16 hi = (__bf16)xs[e];
        vh[e] = hi;
        vl[e] = (__bf16)(xs[e] - (float)hi);
    }
    ((v16bf*)Wh)[idx] = vh;
    ((v16bf*)Wl)[idx] = vl;
}

// ---------------------------------------------------------------------------
// Kernel A: QKV GEMM on pre-packed bf16 fragments + 2D RoPE.
// One wave computes a 16-row x 64-col (one full head) strip of one of
// {Q,K,V}.  grid = (M/16, 3*HEADS).  Hot loop: 10 x b128-pair loads + 12
// v_wmma per k-tile, no conversion VALU.
// ---------------------------------------------------------------------------
__global__ __launch_bounds__(32)
void qkv_rope_kernel(const v16bf* __restrict__ Ah, const v16bf* __restrict__ Al,
                     const v16bf* __restrict__ Bh, const v16bf* __restrict__ Bl,
                     const float* __restrict__ pos,   // (bs, 4096, 2)
                     float* __restrict__ Q, float* __restrict__ K,
                     float* __restrict__ V, int M)
{
    const int lane  = threadIdx.x & 31;
    const int rtile = blockIdx.x;          // 16-row tile
    const int sel   = blockIdx.y;          // 0..23
    const int which = sel >> 3;            // 0=Q 1=K 2=V
    const int h     = sel & 7;             // head

    float* __restrict__ Out = (which == 0) ? Q : (which == 1) ? K : V;
    const int r    = lane & 15;
    const int hi16 = lane >> 4;

    v8f acc[4] = {};                       // cols 0..63 of this head

    #pragma unroll 4
    for (int kt = 0; kt < KT; ++kt) {
        const int aidx = (rtile * KT + kt) * 32 + lane;
        const v16bf ah = Ah[aidx];
        const v16bf al = Al[aidx];
        if (kt + 1 < KT) {
            __builtin_prefetch(&Ah[aidx + 32], 0, 1);   // global_prefetch_b8
            __builtin_prefetch(&Al[aidx + 32], 0, 1);
        }
        #pragma unroll
        for (int t = 0; t < 4; ++t) {
            const int nt   = h * 4 + t;
            const int bidx = ((which * NT + nt) * KT + kt) * 32 + lane;
            const v16bf bh = Bh[bidx];
            const v16bf bl = Bl[bidx];
            // D += Ah*Bh + Ah*Bl + Al*Bh  (~fp32-accurate bf16x3)
            acc[t] = __builtin_amdgcn_wmma_f32_16x16x32_bf16(
                         false, ah, false, bh, (short)0, acc[t], false, false);
            acc[t] = __builtin_amdgcn_wmma_f32_16x16x32_bf16(
                         false, ah, false, bl, (short)0, acc[t], false, false);
            acc[t] = __builtin_amdgcn_wmma_f32_16x16x32_bf16(
                         false, al, false, bh, (short)0, acc[t], false, false);
        }
    }

    // ---- 2D RoPE (Q and K only): C/D layout gives each lane columns
    // c, c+16, c+32, c+48 of the SAME rows -> pure register math. --------
    if (which != 2) {
        const float ifr = __expf(-(float)r * 0.5756462732485114f); // 10000^{-j/16}
        #pragma unroll
        for (int g = 0; g < 8; ++g) {
            const int i  = rtile * 16 + hi16 * 8 + g;     // output node row
            const int b  = i >> 12;                       // batch (N=4096)
            const int n2 = ((i & 511) << 3) + h;          // permuted pos index
            const float* pp = pos + (((size_t)b << 12) + n2) * 2;
            const float ax = pp[0] * 64.0f * ifr;         // t/MIN_FREQ * invfreq
            const float ay = pp[1] * 64.0f * ifr;
            const float cx = __cosf(ax), sx = __sinf(ax);
            const float cy = __cosf(ay), sy = __sinf(ay);
            const float t0 = acc[0][g], t1 = acc[1][g];
            const float t2 = acc[2][g], t3 = acc[3][g];
            acc[0][g] = t0 * cx - t1 * sx;                // d in [0,16)
            acc[1][g] = t1 * cx + t0 * sx;                // d in [16,32)
            acc[2][g] = t2 * cy - t3 * sy;                // d in [32,48)
            acc[3][g] = t3 * cy + t2 * sy;                // d in [48,64)
        }
    }

    // ---- store (node-major (M,512) layout == Qn/Kn/Vn flat) -------------
    #pragma unroll
    for (int t = 0; t < 4; ++t) {
        const int col = h * DHEAD + t * 16 + r;
        #pragma unroll
        for (int g = 0; g < 8; ++g) {
            const int i = rtile * 16 + hi16 * 8 + g;
            Out[(size_t)i * COUT + col] = acc[t][g];
        }
    }
}

// ---------------------------------------------------------------------------
// Zero accumulators (d_out doubles as the wV accumulator) and z.
// ---------------------------------------------------------------------------
__global__ __launch_bounds__(256)
void zero_kernel(float* __restrict__ out, float* __restrict__ z, int M)
{
    const int idx = blockIdx.x * 256 + threadIdx.x;
    if (idx < M * COUT) out[idx] = 0.0f;
    if (idx < M * HEADS) z[idx]  = 0.0f;
}

// ---------------------------------------------------------------------------
// Kernel B: edge attention.  16 lanes per (edge, head): float4 gathers of
// Q[dst]/K[src] (Q/K/V = 96 MB -> resident in the 192 MB L2), width-16
// xor-shuffle dot reduction, clamped exp, f32 scatter-atomics.
// ---------------------------------------------------------------------------
__global__ __launch_bounds__(256)
void edge_attn_kernel(const float* __restrict__ Q,
                      const float* __restrict__ K,
                      const float* __restrict__ V,
                      const int* __restrict__ src,
                      const int* __restrict__ dst,
                      float* __restrict__ out,   // (M,512) wV accumulator
                      float* __restrict__ z,     // (M,8)
                      int E)
{
    const int gid  = blockIdx.x * 256 + threadIdx.x;
    const int pair = gid >> 4;                 // (edge, head) id
    const int subl = gid & 15;                 // lane within the 16-group
    if (pair >= E * HEADS) return;
    const int e = pair >> 3;
    const int h = pair & 7;
    const int s = src[e];
    const int t = dst[e];

    const size_t qoff = (size_t)t * COUT + h * DHEAD + subl * 4;
    const size_t koff = (size_t)s * COUT + h * DHEAD + subl * 4;
    const float4 qv = *(const float4*)(Q + qoff);
    const float4 kv = *(const float4*)(K + koff);
    float p = qv.x * kv.x + qv.y * kv.y + qv.z * kv.z + qv.w * kv.w;
    #pragma unroll
    for (int m = 8; m >= 1; m >>= 1) p += __shfl_xor(p, m, 16);

    const float sc = __expf(fminf(fmaxf(p * 0.125f, -5.0f), 5.0f)); // 1/sqrt(64)

    const float4 vv = *(const float4*)(V + koff);
    float* o = out + (size_t)t * COUT + h * DHEAD + subl * 4;
    atomicAdd(o + 0, vv.x * sc);
    atomicAdd(o + 1, vv.y * sc);
    atomicAdd(o + 2, vv.z * sc);
    atomicAdd(o + 3, vv.w * sc);
    if (subl == 0) atomicAdd(z + (size_t)t * HEADS + h, sc);
}

// ---------------------------------------------------------------------------
// Kernel C: out = wV / z
// ---------------------------------------------------------------------------
__global__ __launch_bounds__(256)
void norm_kernel(float* __restrict__ out, const float* __restrict__ z, int M)
{
    const int idx = blockIdx.x * 256 + threadIdx.x;
    if (idx >= M * COUT) return;
    const int i = idx >> 9;              // node
    const int h = (idx & 511) >> 6;      // head
    out[idx] = out[idx] / z[i * HEADS + h];
}

// ---------------------------------------------------------------------------
extern "C" void kernel_launch(void* const* d_in, const int* in_sizes, int n_in,
                              void* d_out, int out_size, void* d_ws, size_t ws_size,
                              hipStream_t stream)
{
    const float* hin = (const float*)d_in[0];   // (M, 256)
    const float* pos = (const float*)d_in[1];   // (bs, 4096, 2)
    const float* Wq  = (const float*)d_in[2];   // (512, 256)
    const float* Wk  = (const float*)d_in[3];
    const float* Wv  = (const float*)d_in[4];
    const int*   src = (const int*)d_in[5];     // (E,)
    const int*   dst = (const int*)d_in[6];

    const int M = in_sizes[0] / CIN;            // 16384
    const int E = in_sizes[5];                  // 524288

    float* out = (float*)d_out;                 // (M,512) -> doubles as wV acc

    // workspace layout
    float*  Q  = (float*)d_ws;                  // (M,512)
    float*  K  = Q + (size_t)M * COUT;
    float*  V  = K + (size_t)M * COUT;
    float*  z  = V + (size_t)M * COUT;          // (M,8)
    __bf16* Xh = (__bf16*)(z + (size_t)M * HEADS);  // (M,256) packed
    __bf16* Xl = Xh + (size_t)M * CIN;
    __bf16* Wh = Xl + (size_t)M * CIN;          // (3,512,256) packed
    __bf16* Wl = Wh + (size_t)3 * COUT * CIN;

    // 1) pre-pack operands into hi/lo bf16 WMMA fragments (one pass)
    const int axn = (M / 16) * KT * 32;
    pack_x_kernel<<<(axn + 255) / 256, 256, 0, stream>>>(hin, Xh, Xl, M);
    const int bwn = 3 * NT * KT * 32;
    pack_w_kernel<<<(bwn + 255) / 256, 256, 0, stream>>>(Wq, Wk, Wv, Wh, Wl);

    // 2) fused QKV GEMM (bf16x3 WMMA) + RoPE
    dim3 gA(M / 16, 3 * HEADS);
    qkv_rope_kernel<<<gA, 32, 0, stream>>>(
        (const v16bf*)Xh, (const v16bf*)Xl, (const v16bf*)Wh, (const v16bf*)Wl,
        pos, Q, K, V, M);

    // 3) zero accumulators
    zero_kernel<<<(M * COUT + 255) / 256, 256, 0, stream>>>(out, z, M);

    // 4) edge attention (gather / dot / exp / scatter-atomics)
    const int nthreads = E * HEADS * 16;        // 16 lanes per (edge,head)
    edge_attn_kernel<<<(nthreads + 255) / 256, 256, 0, stream>>>(
        Q, K, V, src, dst, out, z, E);

    // 5) normalize
    norm_kernel<<<(M * COUT + 255) / 256, 256, 0, stream>>>(out, z, M);
}
